// TAM_54606214201368
// MI455X (gfx1250) — compile-verified, matrix operand order checked by reading, not run on hardware
//
#include <hip/hip_runtime.h>
#include <cstdint>

typedef __attribute__((ext_vector_type(16))) _Float16 v16h;
typedef __attribute__((ext_vector_type(8)))  _Float16 v8h;
typedef __attribute__((ext_vector_type(8)))  float    v8f;
typedef __attribute__((ext_vector_type(4)))  unsigned int u32x4;
typedef __attribute__((ext_vector_type(8)))  int      i32x8;
typedef __attribute__((ext_vector_type(4)))  int      i32x4;

#define HW 4096
#define CC 256
#define C4 64

#define WMMA_F32_F16(a, b, c) \
  __builtin_amdgcn_wmma_f32_16x16x32_f16(false, (a), false, (b), (short)0, (c), false, false)

// Unified WMMA 16x32 fragment loader (A row-major [m][k], or B via transposed
// storage [n][k]): lane L -> row = L%15? no: row = L&15, K-half = L/16.
// Per-lane halves are two contiguous 16-byte runs -> 2x (ds|global)_load_b128.
// Requires: ld multiple of 8 halves, base 16B-aligned.
static __device__ __forceinline__ v16h ld_frag16(const _Float16* base, int ld, int lane) {
  const int mn = lane & 15, kh = lane >> 4;
  const _Float16* p = base + mn * ld + kh * 8;
  const v8h lo = *(const v8h*)(p);        // k = kh*8 .. kh*8+7
  const v8h hi = *(const v8h*)(p + 16);   // k = 16+kh*8 .. +7
  v16h r;
#pragma unroll
  for (int j = 0; j < 8; ++j) { r[j] = lo[j]; r[j + 8] = hi[j]; }
  return r;
}

// ---- Tensor Data Mover: async 64x64 f16 tile -> LDS (2D descriptor) ----
#if __has_builtin(__builtin_amdgcn_tensor_load_to_lds)
#define HAVE_TDM 1
static __device__ __forceinline__ void tdm_tile64x64(const _Float16* gsrc, _Float16* lds_dst,
                                                     int dim0, int stride0) {
  const unsigned long long ga = (unsigned long long)(uintptr_t)gsrc;
  const unsigned int la = (unsigned int)(uintptr_t)lds_dst;  // flat shared: low 32 = LDS offset
  u32x4 g0;
  g0[0] = 1u;                                                 // count = 1 valid descriptor
  g0[1] = la;                                                 // lds_addr
  g0[2] = (unsigned int)ga;                                   // global_addr[31:0]
  g0[3] = (unsigned int)((ga >> 32) & 0x1FFFFFFu) | (2u << 30);  // addr[56:32] | type=2
  i32x8 g1;
  g1[0] = 1 << 16;                                // workgroup_mask=0, data_size=1 (2B)
  g1[1] = (dim0 & 0xFFFF) << 16;                  // tensor_dim0 lo16
  g1[2] = ((dim0 >> 16) & 0xFFFF) | (64 << 16);   // tensor_dim0 hi16 | tensor_dim1 lo16
  g1[3] = (64 << 16);                             // tensor_dim1 hi16 | tile_dim0=64
  g1[4] = 64;                                     // tile_dim1=64, tile_dim2=0
  g1[5] = stride0;                                // tensor_dim0_stride lo32
  g1[6] = 0;                                      // stride hi16 | dim1_stride lo16
  g1[7] = 0;
  i32x4 z4; z4[0] = z4[1] = z4[2] = z4[3] = 0;    // 2D tensor: groups 2/3 unused
  i32x8 z8;
#pragma unroll
  for (int i = 0; i < 8; ++i) z8[i] = 0;
  __builtin_amdgcn_tensor_load_to_lds(g0, g1, z4, z4, z8, 0);
}
#else
#define HAVE_TDM 0
#endif

// =====================================================================
// Kernel 1: four 1x1-conv projections (GEMM 64x64 tile, K=256), f16 out
//   p=0,1,2: q/kl/kf stored TRANSPOSED [B][HW][64]; p=3: v as [B][64][HW]
// =====================================================================
__global__ __launch_bounds__(128) void proj_kernel(
    const float* __restrict__ last, const float* __restrict__ cur, const float* __restrict__ fut,
    const float* __restrict__ Wqk, const float* __restrict__ bqk,
    const float* __restrict__ Wkl, const float* __restrict__ bkl,
    const float* __restrict__ Wkf, const float* __restrict__ bkf,
    const float* __restrict__ Wv,  const float* __restrict__ bv,
    _Float16* __restrict__ qT, _Float16* __restrict__ klT,
    _Float16* __restrict__ kfT, _Float16* __restrict__ vv)
{
  __shared__ __align__(16) _Float16 Xst[64][40];   // transposed: [hw][k]
  __shared__ __align__(16) _Float16 Ws[64][40];    // [c4][k]
  const int tid = threadIdx.x, lane = tid & 31, w = tid >> 5;
  const int kh = lane >> 4, nl = lane & 15;
  const int hw0 = blockIdx.x * 64;
  const int b = blockIdx.y;
  const int p = blockIdx.z;
  const float* src = (p == 1) ? last : (p == 2) ? fut : cur;
  const float* Wm  = (p == 0) ? Wqk : (p == 1) ? Wkl : (p == 2) ? Wkf : Wv;
  const float* bs  = (p == 0) ? bqk : (p == 1) ? bkl : (p == 2) ? bkf : bv;

  v8f acc[4];
#pragma unroll
  for (int t = 0; t < 4; ++t)
#pragma unroll
    for (int r = 0; r < 8; ++r) acc[t][r] = 0.f;

  for (int k0 = 0; k0 < CC; k0 += 32) {
    __syncthreads();
    for (int i = tid; i < 2048; i += 128) {
      const int r = i >> 6, c = i & 63;                       // r=k row, c=hw col
      Xst[c][r] = (_Float16)src[((long)b * CC + (k0 + r)) * HW + hw0 + c];
      const int r2 = i >> 5, c2 = i & 31;
      Ws[r2][c2] = (_Float16)Wm[r2 * CC + k0 + c2];
    }
    __syncthreads();
    v16h a = ld_frag16(&Ws[16 * w][0], 40, lane);
#pragma unroll
    for (int t = 0; t < 4; ++t) {
      v16h bb = ld_frag16(&Xst[16 * t][0], 40, lane);
      acc[t] = WMMA_F32_F16(a, bb, acc[t]);
    }
  }
#pragma unroll
  for (int t = 0; t < 4; ++t) {
#pragma unroll
    for (int r = 0; r < 8; ++r) {
      const int c4 = 16 * w + r + 8 * kh;
      const int n = hw0 + 16 * t + nl;
      const float val = acc[t][r] + bs[c4];
      if (p == 0)      qT [((long)b * HW + n) * C4 + c4] = (_Float16)val;
      else if (p == 1) klT[((long)b * HW + n) * C4 + c4] = (_Float16)val;
      else if (p == 2) kfT[((long)b * HW + n) * C4 + c4] = (_Float16)val;
      else             vv [((long)b * C4 + c4) * HW + n] = (_Float16)val;
    }
  }
}

// =====================================================================
// Kernel 2: fused flash attention, both temporal directions.
//   Block = 64 queries (4 waves x 16 queries), loops key tiles of 64.
//   K tile (from kT, contiguous 8KB) and V tile DMA'd to LDS by the TDM.
// =====================================================================
__global__ __launch_bounds__(128) void attn_kernel(
    const _Float16* __restrict__ qT, const _Float16* __restrict__ klT,
    const _Float16* __restrict__ kfT, const _Float16* __restrict__ vw,
    _Float16* __restrict__ fused)
{
  __shared__ __align__(16) _Float16 Kt[64 * 64];   // [key][c]   (transposed source)
  __shared__ __align__(16) _Float16 Vs[64 * 64];   // [c][key]
  __shared__ __align__(16) _Float16 Ps[4][16 * 64];// per-wave [q][key]
  __shared__ float red[4][16];

  const int tid = threadIdx.x, lane = tid & 31, w = tid >> 5;
  const int kh = lane >> 4, nl = lane & 15;
  const int b = blockIdx.y, q0 = blockIdx.x * 64;

  const _Float16* qbase = qT + ((long)b * HW + q0 + w * 16) * C4;
  const v16h aq0 = ld_frag16(qbase, C4, lane);        // A(m=q, k=c)  c 0..31
  const v16h aq1 = ld_frag16(qbase + 32, C4, lane);   //               c 32..63
  const _Float16* vsrc = vw + (long)b * C4 * HW;

#pragma unroll 1
  for (int pass = 0; pass < 2; ++pass) {
    const _Float16* kTb = (pass ? kfT : klT) + (long)b * HW * C4;
    v8f acco[4];
    float mrun[8], lrun[8];
#pragma unroll
    for (int mt = 0; mt < 4; ++mt)
#pragma unroll
      for (int r = 0; r < 8; ++r) acco[mt][r] = 0.f;
#pragma unroll
    for (int r = 0; r < 8; ++r) { mrun[r] = -1e30f; lrun[r] = 0.f; }

#pragma unroll 1
    for (int kt = 0; kt < 64; ++kt) {
      const int key0 = kt * 64;
      __syncthreads();                       // prior tile fully consumed
#if HAVE_TDM
      if (w == 0) {
        tdm_tile64x64(kTb + (long)key0 * C4, Kt, C4, C4);  // contiguous [64key][64c]
        tdm_tile64x64(vsrc + key0, Vs, HW, HW);            // [64c][64key]
        __builtin_amdgcn_s_wait_tensorcnt(0);
      }
#else
      for (int i = tid; i < 4096; i += 128) {
        Kt[i] = kTb[(long)key0 * C4 + i];
        Vs[i] = vsrc[(long)(i >> 6) * HW + key0 + (i & 63)];
      }
#endif
      __syncthreads();                       // tiles visible to all waves

      // S[16q x 64key] = Q_tile^T x K_tile  (K-dim = 64 channels)
      v8f s[4];
#pragma unroll
      for (int t = 0; t < 4; ++t) {
#pragma unroll
        for (int r = 0; r < 8; ++r) s[t][r] = 0.f;
        v16h bk0 = ld_frag16(Kt + (16 * t) * 64, 64, lane);       // c 0..31
        s[t] = WMMA_F32_F16(aq0, bk0, s[t]);
        v16h bk1 = ld_frag16(Kt + (16 * t) * 64 + 32, 64, lane);  // c 32..63
        s[t] = WMMA_F32_F16(aq1, bk1, s[t]);
      }

      // online softmax over the key axis
#pragma unroll
      for (int r = 0; r < 8; ++r) {
        float mx = s[0][r];
#pragma unroll
        for (int t = 1; t < 4; ++t) mx = fmaxf(mx, s[t][r]);
#pragma unroll
        for (int d = 1; d < 16; d <<= 1) mx = fmaxf(mx, __shfl_xor(mx, d, 32));
        const float nm = fmaxf(mrun[r], mx);
        const float al = __expf(mrun[r] - nm);
        float sum = 0.f;
#pragma unroll
        for (int t = 0; t < 4; ++t) {
          const float pv = __expf(s[t][r] - nm);
          s[t][r] = pv;
          sum += pv;
        }
#pragma unroll
        for (int d = 1; d < 16; d <<= 1) sum += __shfl_xor(sum, d, 32);
        lrun[r] = lrun[r] * al + sum;
        mrun[r] = nm;
        if (nl == r) red[w][r + 8 * kh] = al;        // per-query rescale factor
#pragma unroll
        for (int t = 0; t < 4; ++t)
          Ps[w][(r + 8 * kh) * 64 + 16 * t + nl] = (_Float16)s[t][r];
      }

      // O[64c x 16q] += V_tile x P^T, with running-max rescale
      const float alq = red[w][nl];
#pragma unroll
      for (int mt = 0; mt < 4; ++mt)
#pragma unroll
        for (int r = 0; r < 8; ++r) acco[mt][r] *= alq;
#pragma unroll
      for (int ks = 0; ks < 2; ++ks) {
        v16h bp = ld_frag16(&Ps[w][32 * ks], 64, lane);   // B(k=key,n=q)=P[q][key]
#pragma unroll
        for (int mt = 0; mt < 4; ++mt) {
          v16h av = ld_frag16(Vs + (16 * mt) * 64 + 32 * ks, 64, lane);
          acco[mt] = WMMA_F32_F16(av, bp, acco[mt]);
        }
      }
    }

    // normalize and write f16 fused[B][128][HW] (pass 0 -> c' 0..63, pass 1 -> 64..127)
#pragma unroll
    for (int r = 0; r < 8; ++r)
      if (nl == r) red[w][r + 8 * kh] = lrun[r];
    const float linv = 1.f / red[w][nl];
#pragma unroll
    for (int mt = 0; mt < 4; ++mt)
#pragma unroll
      for (int r = 0; r < 8; ++r) {
        const int cl = 16 * mt + r + 8 * kh;
        fused[((long)b * 128 + pass * 64 + cl) * HW + q0 + w * 16 + nl] =
            (_Float16)(acco[mt][r] * linv);
      }
    __syncthreads();
  }
}

// =====================================================================
// Kernel 3: fire conv (256x128 GEMM) + folded BatchNorm + residual ReLU
// =====================================================================
__global__ __launch_bounds__(256) void fire_kernel(
    const _Float16* __restrict__ fused, const float* __restrict__ Wf,
    const float* __restrict__ bf, const float* __restrict__ gamma,
    const float* __restrict__ beta, const float* __restrict__ mean,
    const float* __restrict__ var, const float* __restrict__ cur,
    float* __restrict__ out)
{
  __shared__ __align__(16) _Float16 Ws2[128 * 136]; // [o][k], full K=128
  __shared__ __align__(16) _Float16 Fst[64][40];    // transposed: [hw][k]
  __shared__ float scale_s[128], shift_s[128];
  const int tid = threadIdx.x, lane = tid & 31, w = tid >> 5;
  const int kh = lane >> 4, nl = lane & 15;
  const int hw0 = blockIdx.x * 64, b = blockIdx.y, o0 = blockIdx.z * 128;

  if (tid < 128) {
    const int o = o0 + tid;
    const float inv = gamma[o] * rsqrtf(var[o] + 1e-5f);
    scale_s[tid] = inv;
    shift_s[tid] = (bf[o] - mean[o]) * inv + beta[o];
  }
  for (int i = tid; i < 128 * 128; i += 256) {
    const int r = i >> 7, c = i & 127;
    Ws2[r * 136 + c] = (_Float16)Wf[(o0 + r) * 128 + c];
  }
  __syncthreads();

  v8f acc[4];
#pragma unroll
  for (int t = 0; t < 4; ++t)
#pragma unroll
    for (int r = 0; r < 8; ++r) acc[t][r] = 0.f;

  for (int k0 = 0; k0 < 128; k0 += 32) {
    __syncthreads();
    for (int i = tid; i < 2048; i += 256) {
      const int r = i >> 6, c = i & 63;                       // r=k row, c=hw col
      Fst[c][r] = fused[((long)b * 128 + k0 + r) * HW + hw0 + c];
    }
    __syncthreads();
    v16h a = ld_frag16(&Ws2[(16 * w) * 136 + k0], 136, lane);
#pragma unroll
    for (int t = 0; t < 4; ++t) {
      v16h bb = ld_frag16(&Fst[16 * t][0], 40, lane);
      acc[t] = WMMA_F32_F16(a, bb, acc[t]);
    }
  }

#pragma unroll
  for (int t = 0; t < 4; ++t)
#pragma unroll
    for (int r = 0; r < 8; ++r) {
      const int ol = 16 * w + r + 8 * kh;
      const int o = o0 + ol;
      const int hw = hw0 + 16 * t + nl;
      const long idx = ((long)b * CC + o) * HW + hw;
      const float val = acc[t][r] * scale_s[ol] + shift_s[ol];
      out[idx] = fmaxf(cur[idx] + val, 0.f);
    }
}

extern "C" void kernel_launch(void* const* d_in, const int* in_sizes, int n_in,
                              void* d_out, int out_size, void* d_ws, size_t ws_size,
                              hipStream_t stream) {
  (void)in_sizes; (void)n_in; (void)out_size; (void)ws_size;
  const float* last = (const float*)d_in[0];
  const float* cur  = (const float*)d_in[1];
  const float* fut  = (const float*)d_in[2];
  const float* Wqk  = (const float*)d_in[3];
  const float* bqk  = (const float*)d_in[4];
  const float* Wkl  = (const float*)d_in[5];
  const float* bkl  = (const float*)d_in[6];
  const float* Wkf  = (const float*)d_in[7];
  const float* bkf  = (const float*)d_in[8];
  const float* Wv   = (const float*)d_in[9];
  const float* bv   = (const float*)d_in[10];
  const float* Wf   = (const float*)d_in[11];
  const float* bf   = (const float*)d_in[12];
  const float* gm   = (const float*)d_in[13];
  const float* bt   = (const float*)d_in[14];
  const float* mn   = (const float*)d_in[15];
  const float* vr   = (const float*)d_in[16];
  float* out = (float*)d_out;

  _Float16* ws = (_Float16*)d_ws;
  const long B = 4;
  _Float16* qT    = ws;                       // [B][HW][64]
  _Float16* klT   = qT  + B * HW * C4;        // [B][HW][64]
  _Float16* kfT   = klT + B * HW * C4;        // [B][HW][64]
  _Float16* vv    = kfT + B * HW * C4;        // [B][64][HW]
  _Float16* fused = vv  + B * C4 * HW;        // [B][128][HW]

  proj_kernel<<<dim3(64, 4, 4), 128, 0, stream>>>(last, cur, fut, Wqk, bqk, Wkl, bkl,
                                                  Wkf, bkf, Wv, bv, qT, klT, kfT, vv);
  attn_kernel<<<dim3(64, 4, 1), 128, 0, stream>>>(qT, klT, kfT, vv, fused);
  fire_kernel<<<dim3(64, 4, 2), 256, 0, stream>>>(fused, Wf, bf, gm, bt, mn, vr, cur, out);
}
